// RGCNGatherMMSorted_13099650253294
// MI455X (gfx1250) — matrix operand order repaired and checked by reading, not compiled, not run on hardware
//
#include <hip/hip_runtime.h>

typedef __attribute__((ext_vector_type(16))) _Float16 v16h;
typedef __attribute__((ext_vector_type(8)))  _Float16 v8h;
typedef __attribute__((ext_vector_type(8)))  float    v8f;

#define RG_D_IN  128
#define RG_D_OUT 64

// ---------------------------------------------------------------- utilities
__global__ void rg_zero_f32(float* __restrict__ p, int n) {
    int i = blockIdx.x * blockDim.x + threadIdx.x;
    if (i < n) p[i] = 0.0f;
}

// weight [R, K=128, N=64] f32  ->  tw [R, N=64, K=128] f16 (in d_ws)
__global__ void rg_transpose_w(const float* __restrict__ w,
                               _Float16* __restrict__ tw, int total) {
    int i = blockIdx.x * blockDim.x + threadIdx.x;   // linear index into tw
    if (i >= total) return;
    int k  = i & (RG_D_IN - 1);
    int rn = i >> 7;                                  // r*D_OUT + n
    int n  = rn & (RG_D_OUT - 1);
    int r  = rn >> 6;
    tw[i] = (_Float16)w[((size_t)r * RG_D_IN + k) * RG_D_OUT + n];
}

// ---------------------------------------------------------------- main kernel
// One wave32 per 16-edge tile. etypes sorted -> almost all tiles are
// single-relation; mixed tiles loop over runs with inactive A-rows zeroed
// (EXEC stays all-1s across every WMMA, per ISA requirement).
template <bool USE_TW>
__global__ __launch_bounds__(256) void rgcn_gathermm_wmma(
    const float*    __restrict__ feat,    // [N, 128] f32
    const float*    __restrict__ wf,      // [R, 128, 64] f32 (fallback path)
    const _Float16* __restrict__ tw,      // [R, 64, 128] f16 (fast path)
    const int*      __restrict__ src,
    const int*      __restrict__ dst,
    const int*      __restrict__ etypes,
    float*          __restrict__ out,     // [N, 64] f32, pre-zeroed
    int n_edges)
{
    const int lane     = threadIdx.x & 31;
    const int wave     = threadIdx.x >> 5;
    const int tile     = blockIdx.x * (256 >> 5) + wave;
    const int e_base   = tile << 4;
    if (e_base >= n_edges) return;                  // wave-uniform
    const int nrows    = min(16, n_edges - e_base);

    const int row      = lane & 15;                 // A: M row / B,C: N col (local)
    const int half_sel = lane >> 4;                 // 0 or 1
    const int klo      = half_sel << 3;             // unused; A and B use contiguous K offset 16*half_sel

    int my_src = 0, my_dst = 0, my_et = 0x7FFFFFFF;
    if (row < nrows) {
        my_src = src[e_base + row];
        my_dst = dst[e_base + row];
        my_et  = etypes[e_base + row];
    }

    v8f acc[4];
    #pragma unroll
    for (int nt = 0; nt < 4; ++nt)
        #pragma unroll
        for (int j = 0; j < 8; ++j) acc[nt][j] = 0.0f;

    const float* frow = feat + (size_t)my_src * RG_D_IN;

    int done = 0;
    while (done < nrows) {                          // loop over relation runs
        // relation of first unprocessed row (lane `done` holds row `done`)
        const int  r   = __builtin_amdgcn_ds_bpermute(done << 2, my_et);
        const bool act = (my_et == r);
        unsigned gt = __builtin_amdgcn_ballot_w32(my_et > r) & 0xFFFFu;
        const int next = gt ? (int)__builtin_ctz(gt) : 16;

        #pragma unroll
        for (int ks = 0; ks < 4; ++ks) {            // K = 128 in 4 steps of 32
            // ---- A fragment: 16x32, contiguous 16-K per lane half ----
            v16h a;
            #pragma unroll
            for (int j = 0; j < 16; ++j) a[j] = (_Float16)0.0f;
            if (act) {
                const float4* p = (const float4*)(frow + ks * 32 + 16 * half_sel);
                float4 f0 = p[0];
                float4 f1 = p[1];
                const float4* q = (const float4*)(frow + ks * 32 + 16 * half_sel + 8);
                float4 f2 = q[0];
                float4 f3 = q[1];
                a[0]=(_Float16)f0.x;  a[1]=(_Float16)f0.y;
                a[2]=(_Float16)f0.z;  a[3]=(_Float16)f0.w;
                a[4]=(_Float16)f1.x;  a[5]=(_Float16)f1.y;
                a[6]=(_Float16)f1.z;  a[7]=(_Float16)f1.w;
                a[8]=(_Float16)f2.x;  a[9]=(_Float16)f2.y;
                a[10]=(_Float16)f2.z; a[11]=(_Float16)f2.w;
                a[12]=(_Float16)f3.x; a[13]=(_Float16)f3.y;
                a[14]=(_Float16)f3.z; a[15]=(_Float16)f3.w;
            }

            #pragma unroll
            for (int nt = 0; nt < 4; ++nt) {        // N = 64 in 4 tiles of 16
                // ---- B fragment: 32x16, contiguous 16-K per lane half ----
                v16h b;
                if constexpr (USE_TW) {
                    const _Float16* wb = tw + (size_t)r * (RG_D_OUT * RG_D_IN)
                                            + (size_t)(nt * 16 + row) * RG_D_IN
                                            + ks * 32 + 16 * half_sel;
                    v8h b0 = *(const v8h*)wb;       // K +0..7
                    v8h b1 = *(const v8h*)(wb + 8); // K +8..15
                    #pragma unroll
                    for (int j = 0; j < 8; ++j) { b[j] = b0[j]; b[8 + j] = b1[j]; }
                } else {
                    const float* wbf = wf + (size_t)r * (RG_D_IN * RG_D_OUT)
                                          + (size_t)(ks * 32 + 16 * half_sel) * RG_D_OUT
                                          + nt * 16 + row;
                    #pragma unroll
                    for (int j = 0; j < 16; ++j)
                        b[j] = (_Float16)wbf[(size_t)j * RG_D_OUT];
                }

                acc[nt] = __builtin_amdgcn_wmma_f32_16x16x32_f16(
                    /*neg_a=*/false, a, /*neg_b=*/false, b,
                    /*c_mod=*/(short)0, acc[nt],
                    /*reuse_a=*/false, /*reuse_b=*/false);
            }
        }
        done = next;
    }

    // ---- scatter-add: C/D layout lane holds col=row, rows half_sel*8+j ----
    #pragma unroll
    for (int j = 0; j < 8; ++j) {
        const int m  = half_sel * 8 + j;
        const int de = __builtin_amdgcn_ds_bpermute(m << 2, my_dst);
        if (m < nrows) {
            float* orow = out + (size_t)de * RG_D_OUT + row;
            atomicAdd(orow +  0, acc[0][j]);
            atomicAdd(orow + 16, acc[1][j]);
            atomicAdd(orow + 32, acc[2][j]);
            atomicAdd(orow + 48, acc[3][j]);
        }
    }
}

// ---------------------------------------------------------------- launcher
extern "C" void kernel_launch(void* const* d_in, const int* in_sizes, int n_in,
                              void* d_out, int out_size, void* d_ws, size_t ws_size,
                              hipStream_t stream) {
    const float* feat   = (const float*)d_in[0];
    const float* weight = (const float*)d_in[1];
    const int*   src    = (const int*)d_in[2];
    const int*   dst    = (const int*)d_in[3];
    const int*   etypes = (const int*)d_in[4];
    float*       out    = (float*)d_out;

    const int n_edges = in_sizes[2];
    const int w_elems = in_sizes[1];                 // R * 128 * 64
    const int n_tiles = (n_edges + 15) / 16;

    // 1) zero output (we accumulate with atomics; harness does not re-zero)
    {
        int t = 256, b = (out_size + t - 1) / t;
        rg_zero_f32<<<b, t, 0, stream>>>(out, out_size);
    }

    const bool use_tw = ws_size >= (size_t)w_elems * 2;
    _Float16* tw = (_Float16*)d_ws;

    // 2) transpose+convert weights into f16 [R, N, K] scratch (tiny)
    if (use_tw) {
        int t = 256, b = (w_elems + t - 1) / t;
        rg_transpose_w<<<b, t, 0, stream>>>(weight, tw, w_elems);
    }

    // 3) main WMMA gather-mm-scatter: one wave per 16-edge tile, 8 waves/block
    int blocks = (n_tiles + 7) / 8;
    if (use_tw) {
        rgcn_gathermm_wmma<true><<<blocks, 256, 0, stream>>>(
            feat, weight, tw, src, dst, etypes, out, n_edges);
    } else {
        rgcn_gathermm_wmma<false><<<blocks, 256, 0, stream>>>(
            feat, weight, tw, src, dst, etypes, out, n_edges);
    }
}
